// DownSample_24739011624966
// MI455X (gfx1250) — matrix operand order repaired and checked by reading, not compiled, not run on hardware
//
#include <hip/hip_runtime.h>
#include <hip/hip_bf16.h>
#include <math.h>

// ---------------------------------------------------------------------------
// PointNet++ set-abstraction downsample for MI455X (gfx1250, wave32).
// Pipeline: prep(weights->f16, fold BN) -> FPS -> KNN(centers) -> WMMA MLP.
// MLP compute (38 GFLOP) runs on v_wmma_f32_16x16x32_f16 with f32 accumulate.
// ---------------------------------------------------------------------------

typedef __attribute__((ext_vector_type(16))) _Float16 v16h;
typedef __attribute__((ext_vector_type(8)))  _Float16 v8h;
typedef __attribute__((ext_vector_type(8)))  float    v8f;

union V16 { v16h v; v8h h[2]; };

#define NB       4096   // points per batch
#define NBATCH   8
#define NC       1024   // centers
#define KNN_K    32
#define CIN      128
#define K1       131    // cin + 3
#define K1P      160    // padded to 5*32
#define CMID     183
#define N1P      192    // padded to 12*16
#define K2P      192    // padded to 6*32
#define COUT     256

// d_ws byte offsets (all 256-aligned)
#define WS_WP1   0                        // 192*160 f16  = 61440 B
#define WS_WP2   61440                    // 256*192 f16  = 98304 B
#define WS_S1    159744                   // 192 f32
#define WS_T1    160512
#define WS_S2    161280                   // 256 f32
#define WS_T2    162304
#define WS_FPS   163328                   // 8192 int
#define WS_KNN   196096                   // 8*1024*32 int = 1 MB

// ---------------------------------------------------------------------------
// Kernel 0: fold BN into scale/shift, pad + transpose-free f16 weights [N][Kpad]
// ---------------------------------------------------------------------------
__global__ void prep_kernel(const float* __restrict__ W1, const float* __restrict__ b1,
                            const float* __restrict__ g1, const float* __restrict__ be1,
                            const float* __restrict__ m1, const float* __restrict__ v1,
                            const float* __restrict__ W2, const float* __restrict__ b2,
                            const float* __restrict__ g2, const float* __restrict__ be2,
                            const float* __restrict__ m2, const float* __restrict__ v2,
                            _Float16* __restrict__ Wp1, _Float16* __restrict__ Wp2,
                            float* __restrict__ s1, float* __restrict__ t1,
                            float* __restrict__ s2, float* __restrict__ t2) {
    const int i = blockIdx.x * blockDim.x + threadIdx.x;
    const int stride = gridDim.x * blockDim.x;
    for (int idx = i; idx < N1P * K1P; idx += stride) {
        int n = idx / K1P, k = idx % K1P;
        Wp1[idx] = (n < CMID && k < K1) ? (_Float16)W1[n * K1 + k] : (_Float16)0.f;
    }
    for (int idx = i; idx < COUT * K2P; idx += stride) {
        int n = idx / K2P, k = idx % K2P;
        Wp2[idx] = (k < CMID) ? (_Float16)W2[n * CMID + k] : (_Float16)0.f;
    }
    for (int n = i; n < N1P; n += stride) {
        if (n < CMID) {
            float s = g1[n] * rsqrtf(v1[n] + 1e-5f);
            s1[n] = s; t1[n] = (b1[n] - m1[n]) * s + be1[n];
        } else { s1[n] = 0.f; t1[n] = 0.f; }
    }
    for (int n = i; n < COUT; n += stride) {
        float s = g2[n] * rsqrtf(v2[n] + 1e-5f);
        s2[n] = s; t2[n] = (b2[n] - m2[n]) * s + be2[n];
    }
}

// ---------------------------------------------------------------------------
// Kernel 1: farthest point sampling, one block per batch (sequential scan).
// Also writes center_xyz into d_out. Matches jnp.argmax first-index ties.
// ---------------------------------------------------------------------------
__global__ __launch_bounds__(256) void fps_kernel(const float* __restrict__ xyz,
                                                  int* __restrict__ fps_idx,
                                                  float* __restrict__ centers) {
    const int b = blockIdx.x;
    const int tid = threadIdx.x;
    __shared__ float sDist[NB];
    __shared__ float sRv[256];
    __shared__ int   sRi[256];
    const float* xb = xyz + (size_t)b * NB * 3;

    float px[16], py[16], pz[16];
#pragma unroll
    for (int j = 0; j < 16; ++j) {
        int p = tid * 16 + j;
        px[j] = xb[p * 3 + 0]; py[j] = xb[p * 3 + 1]; pz[j] = xb[p * 3 + 2];
        sDist[p] = __builtin_inff();
    }
    int far = 0;
    __syncthreads();

    for (int it = 0; it < NC; ++it) {
        if (tid == 0) {
            fps_idx[b * NC + it] = far;
            centers[(size_t)(b * NC + it) * 3 + 0] = xb[far * 3 + 0];
            centers[(size_t)(b * NC + it) * 3 + 1] = xb[far * 3 + 1];
            centers[(size_t)(b * NC + it) * 3 + 2] = xb[far * 3 + 2];
        }
        const float cx = xb[far * 3 + 0], cy = xb[far * 3 + 1], cz = xb[far * 3 + 2];
        float bv = -1.f; int bi = tid * 16;
#pragma unroll
        for (int j = 0; j < 16; ++j) {
            int p = tid * 16 + j;
            float dx = px[j] - cx, dy = py[j] - cy, dz = pz[j] - cz;
            float d = dx * dx + dy * dy + dz * dz;
            float nd = fminf(sDist[p], d);
            sDist[p] = nd;
            if (nd > bv) { bv = nd; bi = p; }
        }
        sRv[tid] = bv; sRi[tid] = bi;
        __syncthreads();
        for (int s = 128; s > 0; s >>= 1) {
            if (tid < s) {
                float ov = sRv[tid + s]; int oi = sRi[tid + s];
                if (ov > sRv[tid] || (ov == sRv[tid] && oi < sRi[tid])) {
                    sRv[tid] = ov; sRi[tid] = oi;
                }
            }
            __syncthreads();
        }
        far = sRi[0];
        __syncthreads();
    }
}

// ---------------------------------------------------------------------------
// Kernel 2: KNN (k=32) for the FPS centers only — identical to knn-all then
// row-gather. Replace-worst top-32 buffer in LDS, candidate tiles staged in LDS.
// ---------------------------------------------------------------------------
__global__ __launch_bounds__(128) void knn_kernel(const float* __restrict__ xyz,
                                                  const int* __restrict__ fps_idx,
                                                  int* __restrict__ knn_idx) {
    const int tid = threadIdx.x;
    const int b = blockIdx.x >> 3;
    const int qi = (blockIdx.x & 7) * 128 + tid;      // center 0..1023
    __shared__ float tx[128], ty[128], tz[128];
    __shared__ float hd[KNN_K][128];
    __shared__ int   hx[KNN_K][128];

    const float* xb = xyz + (size_t)b * NB * 3;
    const int q = fps_idx[b * NC + qi];
    const float qx = xb[q * 3 + 0], qy = xb[q * 3 + 1], qz = xb[q * 3 + 2];

#pragma unroll
    for (int s = 0; s < KNN_K; ++s) hd[s][tid] = __builtin_inff();
    float worst = __builtin_inff(); int wslot = 0;

    for (int t = 0; t < NB / 128; ++t) {
        int cand = t * 128 + tid;
        __syncthreads();
        tx[tid] = xb[cand * 3 + 0]; ty[tid] = xb[cand * 3 + 1]; tz[tid] = xb[cand * 3 + 2];
        __syncthreads();
        for (int j = 0; j < 128; ++j) {
            float dx = tx[j] - qx, dy = ty[j] - qy, dz = tz[j] - qz;
            float d = dx * dx + dy * dy + dz * dz;
            if (d < worst) {
                hd[wslot][tid] = d; hx[wslot][tid] = t * 128 + j;
                worst = -1.f;
#pragma unroll
                for (int s = 0; s < KNN_K; ++s) {
                    float v = hd[s][tid];
                    if (v > worst) { worst = v; wslot = s; }
                }
            }
        }
    }
    int* o = knn_idx + ((size_t)b * NC + qi) * KNN_K;
#pragma unroll
    for (int s = 0; s < KNN_K; ++s) o[s] = hx[s][tid];
}

// ---------------------------------------------------------------------------
// Kernel 3: gather + MLP1 + MLP2 + maxpool. One wave32 per center.
// A frags gathered straight into registers; B frags read from f16 weights
// in d_ws (L2/WGP$ hot); layer1->layer2 handoff through per-wave LDS slab.
// ---------------------------------------------------------------------------
__device__ inline v8h cvt8(const float* __restrict__ p) {
    const float4* q = (const float4*)p;
    float4 a = q[0], b = q[1];
    v8h r;
    r[0] = (_Float16)a.x; r[1] = (_Float16)a.y; r[2] = (_Float16)a.z; r[3] = (_Float16)a.w;
    r[4] = (_Float16)b.x; r[5] = (_Float16)b.y; r[6] = (_Float16)b.z; r[7] = (_Float16)b.w;
    return r;
}

__global__ __launch_bounds__(128) void mlp_kernel(const float* __restrict__ xyz,
                                                  const float* __restrict__ fea,
                                                  const int* __restrict__ knn_idx,
                                                  const _Float16* __restrict__ Wp1,
                                                  const _Float16* __restrict__ Wp2,
                                                  const float* __restrict__ s1,
                                                  const float* __restrict__ t1,
                                                  const float* __restrict__ s2,
                                                  const float* __restrict__ t2,
                                                  const float* __restrict__ centers,
                                                  float* __restrict__ newfea) {
    __shared__ _Float16 sMid[4][32][200];          // per-wave 32 x 192 (+pad) f16

    const int lane = threadIdx.x & 31;
    const int wave = threadIdx.x >> 5;
    const int gc = blockIdx.x * 4 + wave;          // 0..8191 (b*1024 + c)
    const int b = gc >> 10;
    const bool hiHalf = lane >= 16;
    const int col = lane & 15;
    const int h8 = hiHalf ? 8 : 0;
    const int h16 = hiHalf ? 16 : 0;

    const float cx = centers[(size_t)gc * 3 + 0];
    const float cy = centers[(size_t)gc * 3 + 1];
    const float cz = centers[(size_t)gc * 3 + 2];

    // ---- gather A1 fragments (rows = neighbors, K = [fea(128) | rel(3) | 0]) ----
    V16 A1[2][5];
    const int* nb = knn_idx + (size_t)gc * KNN_K;
#pragma unroll
    for (int mt = 0; mt < 2; ++mt) {
        const int row = mt * 16 + col;             // A layout: lane l -> row l&15
        const int nbr = nb[row];
        const float* fr = fea + ((size_t)b * NB + nbr) * CIN;
#pragma unroll
        for (int kc = 0; kc < 4; ++kc) {
            const int k0 = kc * 32 + h8;
            A1[mt][kc].h[0] = cvt8(fr + k0);
            A1[mt][kc].h[1] = cvt8(fr + k0 + 16);
        }
        // kc=4: relative xyz at k=128..130, zeros beyond
        const float* xr = xyz + ((size_t)b * NB + nbr) * 3;
        const float r0 = xr[0] - cx, r1 = xr[1] - cy, r2 = xr[2] - cz;
        v8h lo, hi;
#pragma unroll
        for (int p = 0; p < 8; ++p) {
            const int k = 128 + h8 + p;
            float v = (k == 128) ? r0 : (k == 129) ? r1 : (k == 130) ? r2 : 0.f;
            lo[p] = (_Float16)v;
            hi[p] = (_Float16)0.f;                 // k+16 >= 144 is always padding
        }
        A1[mt][4].h[0] = lo; A1[mt][4].h[1] = hi;
    }

    // ---- layer 1: [32 x 160] @ [160 x 192] -> BN+ReLU -> f16 LDS ----
    for (int nt = 0; nt < 12; ++nt) {
        const int n = nt * 16 + col;
        const float s = s1[n], t = t1[n];
        v8f acc0 = {}, acc1 = {};
#pragma unroll
        for (int kc = 0; kc < 5; ++kc) {
            const _Float16* wb = Wp1 + (size_t)n * K1P + kc * 32 + h8;
            V16 B; B.h[0] = *(const v8h*)wb; B.h[1] = *(const v8h*)(wb + 16);
            acc0 = __builtin_amdgcn_wmma_f32_16x16x32_f16(false, A1[0][kc].v, false, B.v,
                                                          (short)0, acc0, false, false);
            acc1 = __builtin_amdgcn_wmma_f32_16x16x32_f16(false, A1[1][kc].v, false, B.v,
                                                          (short)0, acc1, false, false);
        }
#pragma unroll
        for (int r = 0; r < 8; ++r) {              // C layout: VGPR r -> M = r (+8 hi)
            float y0 = fmaxf(acc0[r] * s + t, 0.f);
            float y1 = fmaxf(acc1[r] * s + t, 0.f);
            sMid[wave][h8 + r][nt * 16 + col] = (_Float16)y0;
            sMid[wave][16 + h8 + r][nt * 16 + col] = (_Float16)y1;
        }
    }
    asm volatile("s_wait_dscnt 0" ::: "memory");   // wave-local LDS RAW fence

    // ---- reload layer-2 A fragments from LDS slab ----
    V16 A2[2][6];
#pragma unroll
    for (int mt = 0; mt < 2; ++mt) {
        const int row = mt * 16 + col;
#pragma unroll
        for (int kc = 0; kc < 6; ++kc) {
            const int k0 = kc * 32 + h8;
            A2[mt][kc].h[0] = *(const v8h*)&sMid[wave][row][k0];
            A2[mt][kc].h[1] = *(const v8h*)&sMid[wave][row][k0 + 16];
        }
    }

    // ---- layer 2: [32 x 192] @ [192 x 256] -> BN+ReLU -> max over 32 rows ----
    float* ob = newfea + (size_t)gc * COUT;
    for (int nt = 0; nt < 16; ++nt) {
        const int n = nt * 16 + col;
        const float s = s2[n], t = t2[n];
        v8f acc0 = {}, acc1 = {};
#pragma unroll
        for (int kc = 0; kc < 6; ++kc) {
            const _Float16* wb = Wp2 + (size_t)n * K2P + kc * 32 + h8;
            V16 B; B.h[0] = *(const v8h*)wb; B.h[1] = *(const v8h*)(wb + 16);
            acc0 = __builtin_amdgcn_wmma_f32_16x16x32_f16(false, A2[0][kc].v, false, B.v,
                                                          (short)0, acc0, false, false);
            acc1 = __builtin_amdgcn_wmma_f32_16x16x32_f16(false, A2[1][kc].v, false, B.v,
                                                          (short)0, acc1, false, false);
        }
        float m = 0.f;                             // ReLU outputs are >= 0
#pragma unroll
        for (int r = 0; r < 8; ++r) {
            m = fmaxf(m, fmaxf(acc0[r] * s + t, 0.f));
            m = fmaxf(m, fmaxf(acc1[r] * s + t, 0.f));
        }
        m = fmaxf(m, __shfl_xor(m, 16, 32));       // fold lane halves (same column)
        if (!hiHalf) ob[n] = m;
    }
}

// ---------------------------------------------------------------------------
extern "C" void kernel_launch(void* const* d_in, const int* in_sizes, int n_in,
                              void* d_out, int out_size, void* d_ws, size_t ws_size,
                              hipStream_t stream) {
    const float* xyz = (const float*)d_in[0];
    const float* fea = (const float*)d_in[1];
    const float* W1  = (const float*)d_in[2];
    const float* b1  = (const float*)d_in[3];
    const float* g1  = (const float*)d_in[4];
    const float* be1 = (const float*)d_in[5];
    const float* m1  = (const float*)d_in[6];
    const float* v1  = (const float*)d_in[7];
    const float* W2  = (const float*)d_in[8];
    const float* b2  = (const float*)d_in[9];
    const float* g2  = (const float*)d_in[10];
    const float* be2 = (const float*)d_in[11];
    const float* m2  = (const float*)d_in[12];
    const float* v2  = (const float*)d_in[13];

    char* ws = (char*)d_ws;
    _Float16* Wp1 = (_Float16*)(ws + WS_WP1);
    _Float16* Wp2 = (_Float16*)(ws + WS_WP2);
    float* s1 = (float*)(ws + WS_S1);
    float* t1 = (float*)(ws + WS_T1);
    float* s2 = (float*)(ws + WS_S2);
    float* t2 = (float*)(ws + WS_T2);
    int* fpsIdx = (int*)(ws + WS_FPS);
    int* knnIdx = (int*)(ws + WS_KNN);

    float* outCenters = (float*)d_out;                         // [8,1024,3]
    float* outFea = (float*)d_out + (size_t)NBATCH * NC * 3;   // [8,1024,256]

    prep_kernel<<<64, 256, 0, stream>>>(W1, b1, g1, be1, m1, v1,
                                        W2, b2, g2, be2, m2, v2,
                                        Wp1, Wp2, s1, t1, s2, t2);
    fps_kernel<<<NBATCH, 256, 0, stream>>>(xyz, fpsIdx, outCenters);
    knn_kernel<<<NBATCH * 8, 128, 0, stream>>>(xyz, fpsIdx, knnIdx);
    mlp_kernel<<<(NBATCH * NC) / 4, 128, 0, stream>>>(xyz, fea, knnIdx,
                                                      Wp1, Wp2, s1, t1, s2, t2,
                                                      outCenters, outFea);
}
